// GridRNNNaive_47399259079394
// MI455X (gfx1250) — matrix operand order repaired and checked by reading, not compiled, not run on hardware
//
#include <hip/hip_runtime.h>
#include <hip/hip_bf16.h>

// ---------------------------------------------------------------------------
// Grid-RNN (2D tanh grid) for MI455X / gfx1250, wave32 + WMMA bf16 +
// async global->LDS weight staging (ASYNCcnt pipeline).
// DEPTH=2, SRC=TRG=48, B=64, H=256.
// out shape: (2, 48, 48, 2, 64, 256) fp32
// ---------------------------------------------------------------------------

#define SRC_L 48
#define TRG_L 48
#define BD    64
#define HD    256
#define CELL  (BD * HD)          // 16384 floats per (hx or hy) state
#define LDA   264                // padded bf16 LDS row stride for A (256 + 8)
#define LDB   36                 // padded bf16 LDS row stride for B slab (32 + 4)
#define BSLAB (256 * LDB)        // bf16 elements per B K-slab (32 K x 256 N)

typedef __attribute__((ext_vector_type(16))) __bf16 v16bf;
typedef __attribute__((ext_vector_type(8)))  __bf16 v8bf;
typedef __attribute__((ext_vector_type(8)))  float  v8f;
typedef __attribute__((ext_vector_type(4)))  int    v4i;

union bf16x16 { v16bf v; v8bf h[2]; };

#if __has_builtin(__builtin_amdgcn_global_load_async_to_lds_b128)
#define ASYNC_LDS 1
#warning "gfx1250 device pass: using __builtin_amdgcn_global_load_async_to_lds_b128"
#else
#define ASYNC_LDS 0
#warning "async-to-LDS builtin NOT available in this pass; sync staging fallback"
#endif

__device__ __forceinline__ void wait_async_all() {
#if ASYNC_LDS
#if __has_builtin(__builtin_amdgcn_s_wait_asynccnt)
    __builtin_amdgcn_s_wait_asynccnt(0);
#else
    asm volatile("s_wait_asynccnt 0x0" ::: "memory");
#endif
#endif
}

// round-to-nearest-even fp32 -> bf16
__device__ __forceinline__ __bf16 f2bf(float x) {
    unsigned u = __builtin_bit_cast(unsigned, x);
    unsigned r = u + 0x7FFFu + ((u >> 16) & 1u);
    return __builtin_bit_cast(__bf16, (unsigned short)(r >> 16));
}

// Stage a 64x256 fp32 row-major tile (row stride 256) into LDS as bf16,
// padded row stride LDA. 256 threads cooperate.
__device__ __forceinline__ void stage_A(const float* __restrict__ Ag,
                                        __bf16* __restrict__ Abuf, int tid) {
    #pragma unroll 4
    for (int idx = tid; idx < BD * HD; idx += 256) {
        int r = idx >> 8;
        int c = idx & 255;
        Abuf[r * LDA + c] = f2bf(Ag[r * 256 + c]);
    }
}

// Issue staging of one 32-K-wide slab of B (N-major bf16 weights, row
// stride 256) into an LDS slab buffer (row stride LDB).  1024 x 16B chunks,
// 4 per thread, fully coalesced.  Async path uses global_load_async_to_lds.
__device__ __forceinline__ void stage_B_slab(const __bf16* __restrict__ Bt,
                                             __bf16* __restrict__ Bs,
                                             int kb, int tid) {
    #pragma unroll
    for (int p = 0; p < 4; ++p) {
        const int c    = tid + p * 256;     // chunk id 0..1023
        const int n    = c >> 2;            // B row (output column) 0..255
        const int koff = (c & 3) * 8;       // 8 bf16 = 16 bytes per chunk
        const __bf16* gp = Bt + (size_t)n * 256 + kb + koff;
        __bf16*       lp = Bs + n * LDB + koff;
#if ASYNC_LDS
        __builtin_amdgcn_global_load_async_to_lds_b128(
            (v4i*)(void*)gp, (v4i*)(void*)lp, 0, 0);
#else
        *(v8bf*)lp = *(const v8bf*)gp;
#endif
    }
}

// acc[0..7] += A(16x256 slice, LDS) * B(256 x 128 slice, staged via LDS),
// using v_wmma_f32_16x16x32_bf16 with a double-buffered K-slab pipeline:
// slab kt+1 streams global->LDS (ASYNCcnt) while slab kt feeds the WMMAs.
//
// Per-lane fragment layout (wave32, 16-bit operands):
//   A 16x32: row = lane&15 ; lane<16 holds K {k0..k0+7, k0+16..k0+23},
//            lane>=16 holds K {k0+8..k0+15, k0+24..k0+31}
//   B 32x16: col = lane&15 ; same K order as A
__device__ __forceinline__ void gemm_acc(const __bf16* __restrict__ Abuf,
                                         __bf16* __restrict__ Bbuf,   // 2 slabs
                                         const __bf16* __restrict__ Bt,
                                         v8f* acc,
                                         int mtile, int nbase, int lcol, int lhi,
                                         int tid) {
    const int arow = mtile * 16 + lcol;

    stage_B_slab(Bt, Bbuf, 0, tid);          // prologue: slab 0
    wait_async_all();
    __syncthreads();

    #pragma unroll
    for (int kt = 0; kt < 8; ++kt) {
        const __bf16* Bs = Bbuf + (kt & 1) * BSLAB;
        if (kt < 7)                           // overlap: stage slab kt+1
            stage_B_slab(Bt, Bbuf + ((kt + 1) & 1) * BSLAB, (kt + 1) * 32, tid);

        const int k0 = kt * 32 + lhi * 8;     // A per-lane K base
        bf16x16 a;
        a.h[0] = *(const v8bf*)&Abuf[arow * LDA + k0];
        a.h[1] = *(const v8bf*)&Abuf[arow * LDA + k0 + 16];

        const int kboff = lhi * 8;            // B per-lane K base inside slab
        #pragma unroll
        for (int nt = 0; nt < 8; ++nt) {
            const int n = nbase + nt * 16 + lcol;
            bf16x16 b;
            b.h[0] = *(const v8bf*)&Bs[n * LDB + kboff];
            b.h[1] = *(const v8bf*)&Bs[n * LDB + kboff + 16];
            acc[nt] = __builtin_amdgcn_wmma_f32_16x16x32_bf16(
                false, a.v, false, b.v, (short)0, acc[nt], false, false);
        }

        if (kt < 7) {                         // slab kt+1 must be resident
            wait_async_all();
            __syncthreads();
        }
    }
}

// ---------------------------------------------------------------------------
// Weight conversion: Wt[n,k] = bf16(W[k,n]); Uxt/Uyt from U (2H x H).
// ---------------------------------------------------------------------------
__global__ __launch_bounds__(256) void convw_kernel(
    const float* __restrict__ W, const float* __restrict__ U,
    unsigned short* __restrict__ Wt_u, unsigned short* __restrict__ Uxt_u,
    unsigned short* __restrict__ Uyt_u) {
    __bf16* Wt  = (__bf16*)Wt_u;
    __bf16* Uxt = (__bf16*)Uxt_u;
    __bf16* Uyt = (__bf16*)Uyt_u;
    int idx = blockIdx.x * 256 + threadIdx.x;     // 0 .. 65535
    int n = idx >> 8;
    int k = idx & 255;
    Wt [n * 256 + k] = f2bf(W[(size_t)k * 256 + n]);
    Uxt[n * 256 + k] = f2bf(U[(size_t)k * 256 + n]);          // Ux = U[0:256]
    Uyt[n * 256 + k] = f2bf(U[(size_t)(256 + k) * 256 + n]);  // Uy = U[256:512]
}

// ---------------------------------------------------------------------------
// Depth-0 projection: srcW = src @ W ; trgW = trg @ W   (fp32 out)
// ---------------------------------------------------------------------------
__global__ __launch_bounds__(256) void proj_kernel(
    const float* __restrict__ src, const float* __restrict__ trg,
    const unsigned short* __restrict__ Wt_u,
    float* __restrict__ srcW, float* __restrict__ trgW) {
    __shared__ __bf16 Abuf[BD * LDA];
    __shared__ __bf16 Bbuf[2 * BSLAB];
    const __bf16* Wt = (const __bf16*)Wt_u;

    const int blk = blockIdx.x;
    const float* Ag = (blk < SRC_L) ? (src + (size_t)blk * CELL)
                                    : (trg + (size_t)(blk - SRC_L) * CELL);
    float* Og = (blk < SRC_L) ? (srcW + (size_t)blk * CELL)
                              : (trgW + (size_t)(blk - SRC_L) * CELL);

    const int tid  = threadIdx.x;
    const int lane = tid & 31;
    const int wave = tid >> 5;
    const int mtile = wave >> 1;
    const int nbase = (wave & 1) * 128;
    const int lcol  = lane & 15;
    const int lhi   = lane >> 4;

    stage_A(Ag, Abuf, tid);
    __syncthreads();

    v8f acc[8];
    #pragma unroll
    for (int nt = 0; nt < 8; ++nt)
        #pragma unroll
        for (int r = 0; r < 8; ++r) acc[nt][r] = 0.f;

    gemm_acc(Abuf, Bbuf, Wt, acc, mtile, nbase, lcol, lhi, tid);

    #pragma unroll
    for (int nt = 0; nt < 8; ++nt) {
        const int col = nbase + nt * 16 + lcol;
        #pragma unroll
        for (int r = 0; r < 8; ++r) {
            const int row = mtile * 16 + lhi * 8 + r;
            Og[row * 256 + col] = acc[nt][r];
        }
    }
}

// ---------------------------------------------------------------------------
// One grid cell per block.  8 waves, each owns a 16x128 output strip.
// temp = hx_up@Ux + hx_left@Uy + b ; hx = tanh(xw+temp) ; hy = tanh(yw+temp)
//   d==0: xw/yw read from precomputed srcW[i] / trgW[j]
//   d==1: xw = prev_hx@W, yw = prev_hy@W, computed in-cell with WMMA
// ---------------------------------------------------------------------------
__global__ __launch_bounds__(256) void grid_cell_kernel(
    const float* __restrict__ srcW, const float* __restrict__ trgW,
    const unsigned short* __restrict__ Wt_u,
    const unsigned short* __restrict__ Uxt_u,
    const unsigned short* __restrict__ Uyt_u,
    const float* __restrict__ bias,
    float* out, int d, int t, int ilo) {
    __shared__ __bf16 Abuf[BD * LDA];
    __shared__ __bf16 Bbuf[2 * BSLAB];
    const __bf16* Wt  = (const __bf16*)Wt_u;
    const __bf16* Uxt = (const __bf16*)Uxt_u;
    const __bf16* Uyt = (const __bf16*)Uyt_u;

    const int i = ilo + blockIdx.x;
    const int j = t - i;

    const int tid  = threadIdx.x;
    const int lane = tid & 31;
    const int wave = tid >> 5;
    const int mtile = wave >> 1;          // 0..3 -> batch rows 16*mtile..
    const int nbase = (wave & 1) * 128;   // hidden column half
    const int lcol  = lane & 15;
    const int lhi   = lane >> 4;

    // state base offsets in out: (((d*48+i)*48+j)*2 + c) * 16384
    auto state_off = [](int dd, int ii, int jj, int c) -> size_t {
        return ((((size_t)dd * SRC_L + ii) * TRG_L + jj) * 2 + c) * CELL;
    };

    // temp accumulator, initialized with bias broadcast across rows
    v8f acc_t[8];
    #pragma unroll
    for (int nt = 0; nt < 8; ++nt) {
        const float bv = bias[nbase + nt * 16 + lcol];
        #pragma unroll
        for (int r = 0; r < 8; ++r) acc_t[nt][r] = bv;
    }

    if (i > 0) {                                   // + hx_up @ Ux
        stage_A(out + state_off(d, i - 1, j, 0), Abuf, tid);
        __syncthreads();
        gemm_acc(Abuf, Bbuf, Uxt, acc_t, mtile, nbase, lcol, lhi, tid);
        __syncthreads();
    }
    if (j > 0) {                                   // + hx_left @ Uy
        stage_A(out + state_off(d, i, j - 1, 0), Abuf, tid);
        __syncthreads();
        gemm_acc(Abuf, Bbuf, Uyt, acc_t, mtile, nbase, lcol, lhi, tid);
        __syncthreads();
    }

    float* hxp = out + state_off(d, i, j, 0);
    float* hyp = out + state_off(d, i, j, 1);

    if (d == 0) {
        const float* xp = srcW + (size_t)i * CELL;
        const float* yp = trgW + (size_t)j * CELL;
        #pragma unroll
        for (int nt = 0; nt < 8; ++nt) {
            const int col = nbase + nt * 16 + lcol;
            #pragma unroll
            for (int r = 0; r < 8; ++r) {
                const int row = mtile * 16 + lhi * 8 + r;
                const float temp = acc_t[nt][r];
                hxp[row * 256 + col] = tanhf(xp[row * 256 + col] + temp);
                hyp[row * 256 + col] = tanhf(yp[row * 256 + col] + temp);
            }
        }
    } else {
        // xw = prev_hx @ W ; yw = prev_hy @ W (prev layer = depth d-1)
        v8f accx[8], accy[8];
        #pragma unroll
        for (int nt = 0; nt < 8; ++nt)
            #pragma unroll
            for (int r = 0; r < 8; ++r) { accx[nt][r] = 0.f; accy[nt][r] = 0.f; }

        stage_A(out + state_off(d - 1, i, j, 0), Abuf, tid);
        __syncthreads();
        gemm_acc(Abuf, Bbuf, Wt, accx, mtile, nbase, lcol, lhi, tid);
        __syncthreads();

        stage_A(out + state_off(d - 1, i, j, 1), Abuf, tid);
        __syncthreads();
        gemm_acc(Abuf, Bbuf, Wt, accy, mtile, nbase, lcol, lhi, tid);
        __syncthreads();

        #pragma unroll
        for (int nt = 0; nt < 8; ++nt) {
            const int col = nbase + nt * 16 + lcol;
            #pragma unroll
            for (int r = 0; r < 8; ++r) {
                const int row = mtile * 16 + lhi * 8 + r;
                const float temp = acc_t[nt][r];
                hxp[row * 256 + col] = tanhf(accx[nt][r] + temp);
                hyp[row * 256 + col] = tanhf(accy[nt][r] + temp);
            }
        }
    }
}

// ---------------------------------------------------------------------------
// Host orchestration: per depth, convert weights, (depth 0) project inputs,
// then sweep 95 anti-diagonals of the 48x48 grid.
// ---------------------------------------------------------------------------
extern "C" void kernel_launch(void* const* d_in, const int* in_sizes, int n_in,
                              void* d_out, int out_size, void* d_ws, size_t ws_size,
                              hipStream_t stream) {
    (void)in_sizes; (void)n_in; (void)out_size; (void)ws_size;

    const float* src = (const float*)d_in[0];   // (48, 64, 256)
    const float* trg = (const float*)d_in[1];   // (48, 64, 256)
    const float* Ws  = (const float*)d_in[2];   // (2, 256, 256)
    const float* Us  = (const float*)d_in[3];   // (2, 512, 256)
    const float* bs  = (const float*)d_in[4];   // (2, 1, 256)
    float* out = (float*)d_out;                 // (2, 48, 48, 2, 64, 256)

    // workspace layout
    unsigned short* Wt  = (unsigned short*)d_ws;        // 256*256 bf16
    unsigned short* Uxt = Wt  + 256 * 256;
    unsigned short* Uyt = Uxt + 256 * 256;
    float* srcW = (float*)(Uyt + 256 * 256);            // (48, 64, 256) fp32
    float* trgW = srcW + (size_t)SRC_L * CELL;          // (48, 64, 256) fp32

    for (int d = 0; d < 2; ++d) {
        convw_kernel<<<256, 256, 0, stream>>>(
            Ws + (size_t)d * 256 * 256, Us + (size_t)d * 512 * 256, Wt, Uxt, Uyt);

        if (d == 0)
            proj_kernel<<<SRC_L + TRG_L, 256, 0, stream>>>(src, trg, Wt, srcW, trgW);

        for (int t = 0; t <= SRC_L + TRG_L - 2; ++t) {
            const int ilo = (t - (TRG_L - 1) > 0) ? t - (TRG_L - 1) : 0;
            const int ihi = (t < SRC_L - 1) ? t : SRC_L - 1;
            const int ncells = ihi - ilo + 1;
            grid_cell_kernel<<<ncells, 256, 0, stream>>>(
                srcW, trgW, Wt, Uxt, Uyt, bs + (size_t)d * 256, out, d, t, ilo);
        }
    }
}
